// MultiHeadAttention_7395933683839
// MI455X (gfx1250) — matrix operand order repaired and checked by reading, not compiled, not run on hardware
//
#include <hip/hip_runtime.h>

// ---------------------------------------------------------------------------
// MultiHeadAttention for MI455X (gfx1250, wave32, WMMA bf16 16x16x32).
// Pipeline: [QKV proj bf16-GEMM] -> [flash-attention] -> [out proj bf16-GEMM].
// Attention: K/V tiles double-buffered in LDS via async global->LDS copies,
// softmax in exp2 domain, row-max via DPP, row-sum via WMMA against ones.
// ---------------------------------------------------------------------------

#define S_LEN   2048
#define NBATCH  2
#define D_MODEL 1024
#define NHEADS  16
#define DHEAD   64
#define IN_DIM  512
#define BS_ROWS (NBATCH * S_LEN)   // 4096

typedef __attribute__((ext_vector_type(16))) __bf16 v16bf;
typedef __attribute__((ext_vector_type(8)))  float  v8f;

union Frag {
    v16bf v;
    uint4 q[2];
};

// --- bf16 conversions: prefer native cvt instructions --------------------
__device__ __forceinline__ unsigned short bfb(float f) {
    __bf16 h = (__bf16)f;                       // native v_cvt if available
    return __builtin_bit_cast(unsigned short, h);
}
__device__ __forceinline__ unsigned pack2bf(float a, float b) {
#if __has_builtin(__builtin_amdgcn_cvt_pk_bf16_f32)
    typedef __attribute__((ext_vector_type(2))) __bf16 v2bf;
    v2bf v = __builtin_amdgcn_cvt_pk_bf16_f32(a, b);
    return __builtin_bit_cast(unsigned, v);
#else
    return (unsigned)bfb(a) | ((unsigned)bfb(b) << 16);
#endif
}

#if __has_builtin(__builtin_amdgcn_exp2f)
#define EXP2(x) __builtin_amdgcn_exp2f(x)
#else
#define EXP2(x) exp2f(x)
#endif

__device__ __forceinline__ v8f wmma_bf16(const Frag& a, const Frag& b, v8f c) {
    return __builtin_amdgcn_wmma_f32_16x16x32_bf16(
        false, a.v, false, b.v, (short)0, c, false, false);
}

// --- DPP-based max reduction across each 16-lane group (wave32) ----------
template<int CTRL>
__device__ __forceinline__ float dppmax(float x) {
    int y = __builtin_amdgcn_update_dpp(0, __builtin_bit_cast(int, x),
                                        CTRL, 0xF, 0xF, true);
    return fmaxf(x, __builtin_bit_cast(float, y));
}
__device__ __forceinline__ float rowmax16(float x) {
    x = dppmax<0xB1>(x);    // quad_perm [1,0,3,2]  (xor 1)
    x = dppmax<0x4E>(x);    // quad_perm [2,3,0,1]  (xor 2)
    x = dppmax<0x124>(x);   // row_ror:4
    x = dppmax<0x128>(x);   // row_ror:8
    return x;
}

// --- async global->LDS 16-byte copy (CDNA5 ASYNCcnt path), with fallback ---
#if __has_builtin(__builtin_amdgcn_global_load_async_to_lds_b128)
#define HAVE_ASYNC_LDS 1
typedef __attribute__((ext_vector_type(4))) int v4i;
typedef __attribute__((address_space(1))) v4i gv4i_t;   // global int4
typedef __attribute__((address_space(3))) v4i lv4i_t;   // LDS int4
#else
#define HAVE_ASYNC_LDS 0
#endif

__device__ __forceinline__ void async_cp16(const unsigned short* g, unsigned short* l) {
#if HAVE_ASYNC_LDS
    __builtin_amdgcn_global_load_async_to_lds_b128(
        (gv4i_t*)(unsigned long long)g,
        (lv4i_t*)(unsigned int)(unsigned long long)l, 0, 0);
#else
    *(uint4*)l = *(const uint4*)g;
#endif
}
__device__ __forceinline__ void async_fence() {
#if HAVE_ASYNC_LDS
#if __has_builtin(__builtin_amdgcn_s_wait_asynccnt)
    __builtin_amdgcn_s_wait_asynccnt(0);
#else
    asm volatile("s_wait_asynccnt 0" ::: "memory");
#endif
#endif
}

// ---------------------------------------------------------------------------
// Generic bf16 WMMA GEMM:  Out[M,N] = A[M,K] * B[K,N] + bias[N]
//   A_F32:    A operand is f32 in global (convert) vs bf16.
//   OUT_MODE: 0 = bf16 row-major, 1 = bf16 "V-transposed" [b,h,d,s], 2 = f32.
// ---------------------------------------------------------------------------
template<bool A_F32, int OUT_MODE>
__global__ __launch_bounds__(128)
void gemm_bf16_k(const void* __restrict__ Ap, const float* __restrict__ Bp,
                 const float* __restrict__ bias, void* __restrict__ Out,
                 int M, int N, int K) {
    __shared__ __align__(16) unsigned short Asm[64][40];   // [m][k]
    __shared__ __align__(16) unsigned short Bsm[64][40];   // [n][k] (transposed)

    const int tid  = threadIdx.x;
    const int lane = tid & 31;
    const int wave = tid >> 5;
    const int l15  = lane & 15;
    const int half = lane >> 4;
    const int kbA  = half * 8;
    const int kbB  = half * 16;
    const int m0   = blockIdx.y * 64;
    const int n0   = blockIdx.x * 64;

    v8f acc[4] = {};

    for (int k0 = 0; k0 < K; k0 += 32) {
        // --- stage A tile (64 x 32) as bf16 ---
        {
            const int r    = tid >> 1;
            const int koff = (tid & 1) * 16;
            if (A_F32) {
                const float* arow = (const float*)Ap + (size_t)(m0 + r) * K + k0 + koff;
#pragma unroll
                for (int i = 0; i < 4; ++i) {
                    float4 f = *(const float4*)(arow + 4 * i);
                    unsigned lo = pack2bf(f.x, f.y);
                    unsigned hi = pack2bf(f.z, f.w);
                    *(unsigned long long*)&Asm[r][koff + 4 * i] =
                        (unsigned long long)lo | ((unsigned long long)hi << 32);
                }
            } else {
                const unsigned short* arow =
                    (const unsigned short*)Ap + (size_t)(m0 + r) * K + k0 + koff;
                async_cp16(arow,     &Asm[r][koff]);
                async_cp16(arow + 8, &Asm[r][koff + 8]);
            }
        }
        // --- stage B tile (32 x 64) transposed -> Bsm[n][k] ---
        {
            const int kk   = tid >> 2;
            const int noff = (tid & 3) * 16;
            const float* brow = Bp + (size_t)(k0 + kk) * N + n0 + noff;
#pragma unroll
            for (int i = 0; i < 4; ++i) {
                float4 f = *(const float4*)(brow + 4 * i);
                Bsm[noff + 4 * i + 0][kk] = bfb(f.x);
                Bsm[noff + 4 * i + 1][kk] = bfb(f.y);
                Bsm[noff + 4 * i + 2][kk] = bfb(f.z);
                Bsm[noff + 4 * i + 3][kk] = bfb(f.w);
            }
        }
        if (!A_F32) async_fence();
        __syncthreads();

        // --- fragments + 4 WMMAs per wave ---
        Frag a;
        a.q[0] = *(const uint4*)&Asm[wave * 16 + l15][kbA];
        a.q[1] = *(const uint4*)&Asm[wave * 16 + l15][kbA + 16];
#pragma unroll
        for (int t = 0; t < 4; ++t) {
            Frag b;
            b.q[0] = *(const uint4*)&Bsm[t * 16 + l15][kbA];
            b.q[1] = *(const uint4*)&Bsm[t * 16 + l15][kbA + 16];
            acc[t] = wmma_bf16(a, b, acc[t]);
        }
        __syncthreads();
    }

    // --- epilogue: bias + store ---
#pragma unroll
    for (int t = 0; t < 4; ++t) {
        const int col = n0 + t * 16 + l15;
        const float bv = bias[col];
#pragma unroll
        for (int r = 0; r < 8; ++r) {
            const int m = m0 + wave * 16 + r + half * 8;
            const float v = acc[t][r] + bv;
            if (OUT_MODE == 0) {
                ((unsigned short*)Out)[(size_t)m * N + col] = bfb(v);
            } else if (OUT_MODE == 1) {
                const int b = m >> 11, s = m & (S_LEN - 1);
                const int h = col >> 6, d = col & 63;
                ((unsigned short*)Out)[(((size_t)(b * NHEADS + h) * 64 + d) << 11) + s]
                    = bfb(v);
            } else {
                ((float*)Out)[(size_t)m * N + col] = v;
            }
        }
    }
}

// ---------------------------------------------------------------------------
// Flash attention: grid (S/64, H, B), 128 threads (4 waves).
// K/V tiles staged in LDS (double-buffered, async copies); each wave owns 16
// query rows and loops over key tiles of 64 with online softmax.
// ---------------------------------------------------------------------------
__global__ __launch_bounds__(128)
void attn_k(const unsigned short* __restrict__ Qb,
            const unsigned short* __restrict__ Kb,
            const unsigned short* __restrict__ Vt,
            unsigned short* __restrict__ Ctx) {
    __shared__ __align__(16) unsigned short Ksm[2][64][64];  // [buf][key][d]
    __shared__ __align__(16) unsigned short Vsm[2][64][64];  // [buf][d][key]
    __shared__ __align__(16) unsigned short Pl[4][16][64];   // probs per wave

    const int tid  = threadIdx.x;
    const int lane = tid & 31;
    const int wave = tid >> 5;
    const int l15  = lane & 15;
    const int half = lane >> 4;
    const int kbA  = half * 8;
    const int kbB  = half * 16;

    const int b  = blockIdx.z;
    const int h  = blockIdx.y;
    const int q0 = blockIdx.x * 64 + wave * 16;

    const unsigned short* Kbase = Kb + (size_t)b * S_LEN * D_MODEL + h * DHEAD;
    const unsigned short* Vbase = Vt + (size_t)(b * NHEADS + h) * DHEAD * S_LEN;

    // Q fragments (A-operand), D split into two K=32 chunks
    Frag aq0, aq1;
    {
        const unsigned short* qrow =
            Qb + (size_t)(b * S_LEN + q0 + l15) * D_MODEL + h * DHEAD;
        aq0.q[0] = *(const uint4*)(qrow + kbA);
        aq0.q[1] = *(const uint4*)(qrow + kbA + 16);
        aq1.q[0] = *(const uint4*)(qrow + 32 + kbA);
        aq1.q[1] = *(const uint4*)(qrow + 32 + kbA + 16);
    }

    Frag onesF;   // all-ones bf16 B fragment for row sums
    onesF.q[0] = make_uint4(0x3F803F80u, 0x3F803F80u, 0x3F803F80u, 0x3F803F80u);
    onesF.q[1] = onesF.q[0];

    float mrow[8], lrow[8];
#pragma unroll
    for (int r = 0; r < 8; ++r) { mrow[r] = -1e30f; lrow[r] = 0.0f; }
    v8f acc[4] = {};

    // cooperative tile staging: thread t copies 64B of one K row + one V row
    const int sr = tid >> 1;            // 0..63
    const int so = (tid & 1) * 32;      // element offset 0 / 32
    auto stage = [&](int buf, int kt) {
        const unsigned short* kg = Kbase + (size_t)(kt + sr) * D_MODEL + so;
        const unsigned short* vg = Vbase + (size_t)sr * S_LEN + kt + so;
        unsigned short* kl = &Ksm[buf][sr][so];
        unsigned short* vl = &Vsm[buf][sr][so];
#pragma unroll
        for (int i = 0; i < 4; ++i) {
            async_cp16(kg + 8 * i, kl + 8 * i);
            async_cp16(vg + 8 * i, vl + 8 * i);
        }
    };

    stage(0, 0);
    async_fence();
    __syncthreads();

    int cur = 0;
    for (int kt = 0; kt < S_LEN; kt += 64) {
        // issue async copies for the next tile while computing this one
        if (kt + 64 < S_LEN) stage(cur ^ 1, kt + 64);
        // warm L2 two tiles ahead (speculative; OOB dropped silently)
        __builtin_prefetch(Kbase + (size_t)(kt + 128 + l15) * D_MODEL, 0, 3);

        // ---- scores tile 16 x 64 = Q . K^T  (K fragments from LDS) ----
        v8f sc[4] = {};
#pragma unroll
        for (int j = 0; j < 4; ++j) {
            const unsigned short* krow = &Ksm[cur][j * 16 + l15][0];
            Frag b0, b1;
            b0.q[0] = *(const uint4*)(krow + kbA);
            b0.q[1] = *(const uint4*)(krow + kbA + 16);
            b1.q[0] = *(const uint4*)(krow + 32 + kbA);
            b1.q[1] = *(const uint4*)(krow + 32 + kbA + 16);
            sc[j] = wmma_bf16(aq0, b0, sc[j]);
            sc[j] = wmma_bf16(aq1, b1, sc[j]);
        }

        // ---- online softmax in exp2 domain ----
        const float SC2 = 0.18033688f;     // (1/sqrt(64)) * log2(e)
        float alpha8[8];
#pragma unroll
        for (int r = 0; r < 8; ++r) {
            const float s0 = sc[0][r] * SC2;
            const float s1 = sc[1][r] * SC2;
            const float s2 = sc[2][r] * SC2;
            const float s3 = sc[3][r] * SC2;
            float mx = fmaxf(fmaxf(s0, s1), fmaxf(s2, s3));
            mx = rowmax16(mx);
            const float mnew  = fmaxf(mrow[r], mx);
            const float alpha = EXP2(mrow[r] - mnew);
            mrow[r]   = mnew;
            alpha8[r] = alpha;
            const float p0 = EXP2(s0 - mnew);
            const float p1 = EXP2(s1 - mnew);
            const float p2 = EXP2(s2 - mnew);
            const float p3 = EXP2(s3 - mnew);
#pragma unroll
            for (int t = 0; t < 4; ++t) acc[t][r] *= alpha;
            const int row = r + half * 8;
            Pl[wave][row][l15]      = bfb(p0);
            Pl[wave][row][l15 + 16] = bfb(p1);
            Pl[wave][row][l15 + 32] = bfb(p2);
            Pl[wave][row][l15 + 48] = bfb(p3);
        }
        asm volatile("s_wait_dscnt 0" ::: "memory");   // intra-wave LDS RAW

        // ---- P (A-operand) fragments from LDS ----
        Frag aP0, aP1;
        aP0.q[0] = *(const uint4*)&Pl[wave][l15][kbA];
        aP0.q[1] = *(const uint4*)&Pl[wave][l15][kbA + 16];
        aP1.q[0] = *(const uint4*)&Pl[wave][l15][32 + kbA];
        aP1.q[1] = *(const uint4*)&Pl[wave][l15][32 + kbA + 16];

        // ---- row-sum of P on the matrix pipe ----
        v8f ls = {};
        ls = wmma_bf16(aP0, onesF, ls);
        ls = wmma_bf16(aP1, onesF, ls);
#pragma unroll
        for (int r = 0; r < 8; ++r) lrow[r] = lrow[r] * alpha8[r] + ls[r];

        // ---- ctx += P . V  (V fragments from LDS) ----
#pragma unroll
        for (int t = 0; t < 4; ++t) {
            const unsigned short* vcol = &Vsm[cur][t * 16 + l15][0];
            Frag bv0, bv1;
            bv0.q[0] = *(const uint4*)(vcol + kbA);
            bv0.q[1] = *(const uint4*)(vcol + kbA + 16);
            bv1.q[0] = *(const uint4*)(vcol + 32 + kbA);
            bv1.q[1] = *(const uint4*)(vcol + 32 + kbA + 16);
            acc[t] = wmma_bf16(aP0, bv0, acc[t]);
            acc[t] = wmma_bf16(aP1, bv1, acc[t]);
        }

        async_fence();      // next tile landed in LDS
        __syncthreads();    // everyone done reading cur / writing next
        cur ^= 1;
    }

    // ---- epilogue: normalize and store bf16 ctx [BS, D_MODEL] ----
#pragma unroll
    for (int t = 0; t < 4; ++t) {
#pragma unroll
        for (int r = 0; r < 8; ++r) {
            const int row = q0 + r + half * 8;
            const float v = acc[t][r] / lrow[r];
            Ctx[(size_t)(b * S_LEN + row) * D_MODEL + h * DHEAD + t * 16 + l15]
                = bfb(v);
        }
    }
}

// ---------------------------------------------------------------------------
extern "C" void kernel_launch(void* const* d_in, const int* in_sizes, int n_in,
                              void* d_out, int out_size, void* d_ws, size_t ws_size,
                              hipStream_t stream) {
    const float* q  = (const float*)d_in[0];
    const float* k  = (const float*)d_in[1];
    const float* v  = (const float*)d_in[2];
    const float* Wq = (const float*)d_in[3];
    const float* bq = (const float*)d_in[4];
    const float* Wk = (const float*)d_in[5];
    const float* bk = (const float*)d_in[6];
    const float* Wv = (const float*)d_in[7];
    const float* bv = (const float*)d_in[8];
    const float* Wo = (const float*)d_in[9];
    const float* bo = (const float*)d_in[10];
    float* out = (float*)d_out;

    const size_t buf = (size_t)BS_ROWS * D_MODEL;   // bf16 elements (8 MB each)
    unsigned short* Qb  = (unsigned short*)d_ws;
    unsigned short* Kb  = Qb + buf;
    unsigned short* Vt  = Kb + buf;
    unsigned short* Ctx = Vt + buf;

    dim3 blk(128);

    dim3 g1(D_MODEL / 64, BS_ROWS / 64);
    gemm_bf16_k<true, 0><<<g1, blk, 0, stream>>>(q, Wq, bq, Qb, BS_ROWS, D_MODEL, IN_DIM);
    gemm_bf16_k<true, 0><<<g1, blk, 0, stream>>>(k, Wk, bk, Kb, BS_ROWS, D_MODEL, IN_DIM);
    gemm_bf16_k<true, 1><<<g1, blk, 0, stream>>>(v, Wv, bv, Vt, BS_ROWS, D_MODEL, IN_DIM);

    dim3 g2(S_LEN / 64, NHEADS, NBATCH);
    attn_k<<<g2, blk, 0, stream>>>(Qb, Kb, Vt, Ctx);

    dim3 g3(IN_DIM / 64, BS_ROWS / 64);
    gemm_bf16_k<false, 2><<<g3, blk, 0, stream>>>(Ctx, Wo, bo, out, BS_ROWS, IN_DIM, D_MODEL);
}
